// MinGRU_43430709297203
// MI455X (gfx1250) — matrix operand order repaired and checked by reading, not compiled, not run on hardware
//
#include <hip/hip_runtime.h>
#include <hip/hip_bf16.h>

// Problem dims (fixed by setup_inputs)
constexpr int Bd = 4, Td = 8192, Dd = 512, Hd = 512;
constexpr int BT = Bd * Td;                  // 32768 rows
constexpr int CH = 64;                       // scan chunks per sequence
constexpr int CL = Td / CH;                  // 128 steps per chunk

constexpr int NT = 64;                       // N-tile cols per block
constexpr unsigned LDS_BYTES = 2u * NT * Dd * 2u;   // both gates: 128 KB

typedef __attribute__((ext_vector_type(16))) __bf16        v16bf;
typedef __attribute__((ext_vector_type(8)))  float         v8f;
typedef __attribute__((ext_vector_type(8)))  unsigned int  v8u;

union Frag { v8u u; v16bf bf; };

// ---------- f32 -> bf16 (round to nearest even), packed pair ----------
__device__ __forceinline__ unsigned int f32_to_bf16_bits(float f) {
    unsigned int u = __float_as_uint(f);
    u += 0x7FFFu + ((u >> 16) & 1u);
    return u >> 16;
}

__global__ void cvt_f32_to_bf16_pk(const float* __restrict__ src,
                                   unsigned int* __restrict__ dst, int n2) {
    int i = blockIdx.x * blockDim.x + threadIdx.x;
    if (i < n2) {
        float2 v = reinterpret_cast<const float2*>(src)[i];
        dst[i] = f32_to_bf16_bits(v.x) | (f32_to_bf16_bits(v.y) << 16);
    }
}

// ---------- stable helpers ----------
__device__ __forceinline__ float softplus_f(float t) {      // log(1+exp(t))
    return fmaxf(t, 0.0f) + log1pf(__expf(-fabsf(t)));
}

__device__ __forceinline__ float logaddexp_f(float p, float q) {
    float mm = fmaxf(p, q);
    return mm + __logf(__expf(p - mm) + __expf(q - mm));
}

// ---------- fused dual-GEMM + gate epilogue ----------
// Block = 8 waves sharing one 64-col N tile; wave w owns rows [mBase, mBase+16).
// W slice for both gates (128 KB) is staged Global->LDS once via CDNA5 async
// copy (ASYNCcnt), then B fragments are two 16B ds_load_b128 per fragment.
// A (16x32 bf16): lanes 0-15 = M rows with K 0..7 & 16..23; lanes 16-31 same M,
// K 8..15 & 24..31.  B (32x16): lanes 0-15 col N K 0..7 & 16..23; lanes 16-31 K 8..15 & 24..31.
__global__ __launch_bounds__(256) void mingru_gemm_gates(
    const unsigned short* __restrict__ xbf,
    const unsigned short* __restrict__ wzbf,
    const unsigned short* __restrict__ whbf,
    const float* __restrict__ bz, const float* __restrict__ bh,
    float* __restrict__ ws_lc,     // log(1 - z)  (B*T, H)
    float* __restrict__ out)       // log z + log g(a)  staged into d_out
{
    extern __shared__ unsigned short smem[];   // [gate][col][k] : 2*64*512 bf16

    const int tid   = threadIdx.x;
    const int lane  = tid & 31;
    const int wv    = tid >> 5;                 // wave in block: 0..7
    const int nblk  = blockIdx.x & 7;           // 8 N tiles of 64 cols
    const int mblk  = blockIdx.x >> 3;          // 256 M blocks of 128 rows
    const int mBase = mblk * 128 + wv * 16;
    const int nBase = nblk * NT;
    const int laneM = lane & 15;
    const int hiH   = lane >> 4;                // lane half select

    // ---- stage W slice (both gates) into LDS with async copy ----
    // 128 KB = 8192 x 16B chunks; 256 threads x 32 iterations.
    #pragma unroll 4
    for (int it = 0; it < 32; ++it) {
        const int i   = tid + it * 256;         // chunk id 0..8191
        const int g   = i >> 12;                // gate select
        const int rem = i & 4095;
        const int c   = rem >> 6;               // col within tile
        const int kc  = rem & 63;               // 16B chunk along K
        const unsigned short* src = g ? whbf : wzbf;
        const unsigned short* gp  = src + (size_t)(nBase + c) * Dd + kc * 8;
        const unsigned int ldsoff = (unsigned int)i * 16u;
        asm volatile("global_load_async_to_lds_b128 %0, %1, off"
                     :: "v"(ldsoff), "v"(gp)
                     : "memory");
    }
    asm volatile("s_wait_asynccnt 0x0" ::: "memory");
    __syncthreads();

    v8f accz[4] = {v8f{0.f,0.f,0.f,0.f,0.f,0.f,0.f,0.f},
                   v8f{0.f,0.f,0.f,0.f,0.f,0.f,0.f,0.f},
                   v8f{0.f,0.f,0.f,0.f,0.f,0.f,0.f,0.f},
                   v8f{0.f,0.f,0.f,0.f,0.f,0.f,0.f,0.f}};
    v8f acch[4] = {accz[0], accz[1], accz[2], accz[3]};

    const unsigned short* aRow = xbf + (size_t)(mBase + laneM) * Dd;

    for (int kk = 0; kk < Dd; kk += 32) {
        // A fragment: 2x global b128 per lane
        Frag a;
        const int ka = kk + hiH * 8;
        #pragma unroll
        for (int j = 0; j < 4; ++j)
            a.u[j]     = *reinterpret_cast<const unsigned int*>(aRow + ka + 2 * j);
        #pragma unroll
        for (int j = 0; j < 4; ++j)
            a.u[4 + j] = *reinterpret_cast<const unsigned int*>(aRow + ka + 16 + 2 * j);

        // B fragments from LDS: per fragment two 16B (aligned) ds_load_b128
        const int kb = kk + hiH * 8;
        Frag bzf[4], bhf[4];
        #pragma unroll
        for (int t = 0; t < 4; ++t) {
            const unsigned short* p =
                &smem[(size_t)(t * 16 + laneM) * Dd + kb];
            const uint4 lo = *reinterpret_cast<const uint4*>(p);
            const uint4 hi = *reinterpret_cast<const uint4*>(p + 16);
            bzf[t].u[0] = lo.x; bzf[t].u[1] = lo.y;
            bzf[t].u[2] = lo.z; bzf[t].u[3] = lo.w;
            bzf[t].u[4] = hi.x; bzf[t].u[5] = hi.y;
            bzf[t].u[6] = hi.z; bzf[t].u[7] = hi.w;
        }
        #pragma unroll
        for (int t = 0; t < 4; ++t) {
            const unsigned short* p =
                &smem[(size_t)(NT * Dd) + (size_t)(t * 16 + laneM) * Dd + kb];
            const uint4 lo = *reinterpret_cast<const uint4*>(p);
            const uint4 hi = *reinterpret_cast<const uint4*>(p + 16);
            bhf[t].u[0] = lo.x; bhf[t].u[1] = lo.y;
            bhf[t].u[2] = lo.z; bhf[t].u[3] = lo.w;
            bhf[t].u[4] = hi.x; bhf[t].u[5] = hi.y;
            bhf[t].u[6] = hi.z; bhf[t].u[7] = hi.w;
        }
        #pragma unroll
        for (int t = 0; t < 4; ++t)
            accz[t] = __builtin_amdgcn_wmma_f32_16x16x32_bf16(
                false, a.bf, false, bzf[t].bf, (short)0, accz[t], false, false);
        #pragma unroll
        for (int t = 0; t < 4; ++t)
            acch[t] = __builtin_amdgcn_wmma_f32_16x16x32_bf16(
                false, a.bf, false, bhf[t].bf, (short)0, acch[t], false, false);
    }

    // Epilogue: C layout — VGPR r: M = r + 8*hiH, N = laneM.  Branchless log_g.
    #pragma unroll
    for (int t = 0; t < 4; ++t) {
        const int h   = nBase + t * 16 + laneM;
        const float bzv = bz[h];
        const float bhv = bh[h];
        #pragma unroll
        for (int r = 0; r < 8; ++r) {
            const int m = mBase + r + 8 * hiH;
            const float kv = accz[t][r] + bzv;
            const float av = acch[t][r] + bhv;
            const float lc   = -softplus_f(kv);           // log(1 - sigmoid(k))
            const float lz   = -softplus_f(-kv);          // log sigmoid(k)
            const float lpos = __logf(av + 0.5f);
            const float lneg = -softplus_f(-av);
            const float lth  = (av >= 0.0f) ? lpos : lneg;   // log g(a)
            const size_t idx = (size_t)m * Hd + h;
            ws_lc[idx] = lc;
            out[idx]   = lz + lth;
        }
    }
}

// ---------- chunked log-space scan ----------
// Recurrence lh <- logaddexp(lc + lh, lv) is composition of ops s=(a,b):
// lh -> logaddexp(a + lh, b);  s2*s1 = (a2+a1, logaddexp(a2+b1, b2)).

__global__ void scan_phase1(const float* __restrict__ ws_lc,
                            const float* __restrict__ lv_buf,
                            float* __restrict__ Aarr,
                            float* __restrict__ Barr) {
    int tid = blockIdx.x * blockDim.x + threadIdx.x;   // (b, c, h) flat
    if (tid >= Bd * CH * Hd) return;
    const int h    = tid % Hd;
    const int rest = tid / Hd;
    const int c    = rest % CH;
    const int b    = rest / CH;
    size_t idx = ((size_t)b * Td + (size_t)c * CL) * Hd + h;
    float Asum = 0.0f;
    float v    = -1.0e30f;                 // finite stand-in for -inf
    for (int t = 0; t < CL; ++t, idx += Hd) {
        const float lc = ws_lc[idx];
        const float lv = lv_buf[idx];
        Asum += lc;
        v = logaddexp_f(lc + v, lv);
    }
    Aarr[tid] = Asum;
    Barr[tid] = v;
}

__global__ void scan_phase2(const float* __restrict__ h_prev,
                            const float* __restrict__ Aarr,
                            const float* __restrict__ Barr,
                            float* __restrict__ Sarr) {
    int tid = blockIdx.x * blockDim.x + threadIdx.x;
    if (tid >= Bd * Hd) return;
    const int b = tid / Hd;
    const int h = tid - b * Hd;
    float s = __logf(h_prev[b * Hd + h]);
    #pragma unroll 4
    for (int c = 0; c < CH; ++c) {
        const size_t i = ((size_t)b * CH + c) * Hd + h;
        Sarr[i] = s;
        s = logaddexp_f(Aarr[i] + s, Barr[i]);
    }
}

__global__ void scan_phase3(const float* __restrict__ ws_lc,
                            const float* __restrict__ Sarr,
                            float* __restrict__ out) {
    int tid = blockIdx.x * blockDim.x + threadIdx.x;   // (b, c, h) flat
    if (tid >= Bd * CH * Hd) return;
    const int h    = tid % Hd;
    const int rest = tid / Hd;
    const int c    = rest % CH;
    const int b    = rest / CH;
    float lh = Sarr[tid];
    size_t idx = ((size_t)b * Td + (size_t)c * CL) * Hd + h;
    for (int t = 0; t < CL; ++t, idx += Hd) {
        const float lc = ws_lc[idx];
        const float lv = out[idx];
        lh = logaddexp_f(lc + lh, lv);
        out[idx] = __expf(lh);
    }
}

extern "C" void kernel_launch(void* const* d_in, const int* in_sizes, int n_in,
                              void* d_out, int out_size, void* d_ws, size_t ws_size,
                              hipStream_t stream) {
    const float* x      = (const float*)d_in[0];   // (B,T,D)
    const float* h_prev = (const float*)d_in[1];   // (B,1,H)
    const float* W_h    = (const float*)d_in[2];   // (H,D)
    const float* b_h    = (const float*)d_in[3];   // (H)
    const float* W_z    = (const float*)d_in[4];   // (H,D)
    const float* b_z    = (const float*)d_in[5];   // (H)
    float* out = (float*)d_out;

    char* ws = (char*)d_ws;
    size_t off = 0;
    unsigned short* xbf  = (unsigned short*)(ws + off); off += (size_t)BT * Dd * 2;   // 32 MB
    unsigned short* wzbf = (unsigned short*)(ws + off); off += (size_t)Hd * Dd * 2;   // 512 KB
    unsigned short* whbf = (unsigned short*)(ws + off); off += (size_t)Hd * Dd * 2;   // 512 KB
    float* ws_lc = (float*)(ws + off); off += (size_t)BT * Hd * 4;                    // 64 MB
    float* Aarr  = (float*)(ws + off); off += (size_t)Bd * CH * Hd * 4;               // 512 KB
    float* Barr  = (float*)(ws + off); off += (size_t)Bd * CH * Hd * 4;               // 512 KB
    float* Sarr  = (float*)(ws + off); off += (size_t)Bd * CH * Hd * 4;               // 512 KB

    // Allow 128 KB dynamic LDS (gfx1250 WGP has 320 KB)
    (void)hipFuncSetAttribute((const void*)mingru_gemm_gates,
                              hipFuncAttributeMaxDynamicSharedMemorySize,
                              (int)LDS_BYTES);

    // 1) precision staging: f32 -> bf16
    {
        int n2 = (BT * Dd) / 2;
        cvt_f32_to_bf16_pk<<<(n2 + 255) / 256, 256, 0, stream>>>(
            x, (unsigned int*)xbf, n2);
        int w2 = (Hd * Dd) / 2;
        cvt_f32_to_bf16_pk<<<(w2 + 255) / 256, 256, 0, stream>>>(
            W_z, (unsigned int*)wzbf, w2);
        cvt_f32_to_bf16_pk<<<(w2 + 255) / 256, 256, 0, stream>>>(
            W_h, (unsigned int*)whbf, w2);
    }

    // 2) fused dual-GEMM (WMMA bf16, f32 accumulate) + gate epilogue
    //    2048 blocks of 8 waves; a block's waves share the LDS-staged W slice.
    mingru_gemm_gates<<<2048, 256, LDS_BYTES, stream>>>(xbf, wzbf, whbf,
                                                        b_z, b_h, ws_lc, out);

    // 3) chunked log-space scan: 64x shorter dependent chains than naive scan
    {
        int n1 = Bd * CH * Hd;                         // 131072 threads
        scan_phase1<<<(n1 + 255) / 256, 256, 0, stream>>>(ws_lc, out, Aarr, Barr);
        int n2 = Bd * Hd;                              // 2048 threads
        scan_phase2<<<(n2 + 255) / 256, 256, 0, stream>>>(h_prev, Aarr, Barr, Sarr);
        scan_phase3<<<(n1 + 255) / 256, 256, 0, stream>>>(ws_lc, Sarr, out);
    }
}
